// StackedLoRALinear_35270271435032
// MI455X (gfx1250) — compile-verified, matrix operand order checked by reading, not run on hardware
//
#include <hip/hip_runtime.h>
#include <hip/hip_bf16.h>

// ---------------------------------------------------------------------------
// StackedLoRALinear for MI455X (gfx1250):
//   y = x @ (W + sum_d B_d A_d)^T + b       (SCALE = 1)
// Kernel 1: fold the rank-128 LoRA update into W, emit bf16 hi/lo split.
// Kernel 2: 3xBF16-split GEMM on v_wmma_f32_16x16x32_bf16 with fused bias.
//   Block tile 128x128, BK=32, 8 wave32s (wave tile 32x64 = 2x4 frags).
// ---------------------------------------------------------------------------

typedef __bf16 bf16_t;
typedef __attribute__((ext_vector_type(16))) __bf16 v16bf;
typedef __attribute__((ext_vector_type(8)))  __bf16 v8bf;
typedef __attribute__((ext_vector_type(8)))  float  v8f;

#define D_DIM 3072
#define M_DIM 16384       // B*S = 4*4096
#define LORA_DEPTH 4
#define LORA_RANK 32

#define BM 128
#define BN 128
#define BK 32
#define MFRAG 2           // wave tile M = 32
#define NFRAG 4           // wave tile N = 64

// ---------------------------------------------------------------------------
// Kernel 1: W_eff[o,i] = W[o,i] + sum_{d,r} lora_B[d,o,r] * lora_A[d,r,i]
// then split into bf16 hi/lo. lora_B accesses are wave-uniform (scalar loads),
// lora_A accesses are fully coalesced over i and L2-resident (1.5 MB).
// ---------------------------------------------------------------------------
__global__ __launch_bounds__(256) void build_weff_kernel(
    const float* __restrict__ W,
    const float* __restrict__ lA,   // [depth, rank, D]
    const float* __restrict__ lB,   // [depth, D, rank]
    bf16_t* __restrict__ whi,
    bf16_t* __restrict__ wlo)
{
    const int i = blockIdx.x * 256 + threadIdx.x;
    const int o = blockIdx.y;

    float acc = W[(size_t)o * D_DIM + i];
#pragma unroll
    for (int d = 0; d < LORA_DEPTH; ++d) {
        const float* __restrict__ Bd = lB + ((size_t)d * D_DIM + o) * LORA_RANK;
        const float* __restrict__ Ad = lA + (size_t)d * LORA_RANK * D_DIM + i;
#pragma unroll
        for (int r = 0; r < LORA_RANK; ++r) {
            acc = fmaf(Bd[r], Ad[(size_t)r * D_DIM], acc);
        }
    }
    bf16_t hi = (bf16_t)acc;
    bf16_t lo = (bf16_t)(acc - (float)hi);
    whi[(size_t)o * D_DIM + i] = hi;
    wlo[(size_t)o * D_DIM + i] = lo;
}

// ---------------------------------------------------------------------------
// Kernel 2: out[M,N] = X[M,K] @ Weff[N,K]^T + bias, 3xBF16 split WMMA.
// ---------------------------------------------------------------------------
__global__ __launch_bounds__(256) void lora_gemm_kernel(
    const float*  __restrict__ X,     // [M, K] fp32
    const bf16_t* __restrict__ Whi,   // [N, K] bf16 (hi part)
    const bf16_t* __restrict__ Wlo,   // [N, K] bf16 (lo part)
    const float*  __restrict__ bias,  // [N]
    float*        __restrict__ out,   // [M, N]
    int M, int N, int K)
{
    __shared__ __align__(64) bf16_t sAhi[2][BM * BK];   // 8 KB per buffer
    __shared__ __align__(64) bf16_t sAlo[2][BM * BK];
    __shared__ __align__(64) bf16_t sBhi[2][BN * BK];   // 8 KB per buffer
    __shared__ __align__(64) bf16_t sBlo[2][BN * BK];   // total LDS = 64 KB

    const int tid      = threadIdx.x;
    const int lane     = tid & 31;
    const int wave     = tid >> 5;
    const int laneRow  = lane & 15;
    const int laneHalf = lane >> 4;           // selects K half / M half

    const int mWave = (wave >> 1) * (MFRAG * 16);   // 0,32,64,96
    const int nWave = (wave & 1) * (NFRAG * 16);    // 0,64

    const int blockN0 = blockIdx.x * BN;
    const int blockM0 = blockIdx.y * BM;

    // Global->LDS mapping (256 threads):
    //   A tile 128x32 f32: 2 threads/row, 16 floats each (4x float4)
    const int aRow = tid >> 1;
    const int aCol = (tid & 1) * 16;
    //   B tiles 128x32 bf16: 2 threads/row, 16 bf16 each (2x 16B per array)
    const int bRow = tid >> 1;
    const int bCol = (tid & 1) * 16;

    const int nk = K / BK;                    // 96 K-steps

    v8f acc[MFRAG][NFRAG];
#pragma unroll
    for (int mf = 0; mf < MFRAG; ++mf)
#pragma unroll
        for (int nf = 0; nf < NFRAG; ++nf)
            acc[mf][nf] = (v8f){0.f, 0.f, 0.f, 0.f, 0.f, 0.f, 0.f, 0.f};

    auto load_tile = [&](int kt, int buf) {
        // ---- A: load fp32 x, split into bf16 hi/lo on the fly ----
        const float* __restrict__ src =
            X + (size_t)(blockM0 + aRow) * K + (size_t)kt * BK + aCol;
        float f[16];
        const float4* sv = (const float4*)src;
#pragma unroll
        for (int q = 0; q < 4; ++q) {
            float4 v = sv[q];
            f[q * 4 + 0] = v.x; f[q * 4 + 1] = v.y;
            f[q * 4 + 2] = v.z; f[q * 4 + 3] = v.w;
        }
        v8bf hi0, hi1, lo0, lo1;
#pragma unroll
        for (int j = 0; j < 8; ++j) {
            bf16_t h = (bf16_t)f[j];
            hi0[j] = h;
            lo0[j] = (bf16_t)(f[j] - (float)h);
        }
#pragma unroll
        for (int j = 0; j < 8; ++j) {
            bf16_t h = (bf16_t)f[8 + j];
            hi1[j] = h;
            lo1[j] = (bf16_t)(f[8 + j] - (float)h);
        }
        *(v8bf*)&sAhi[buf][aRow * BK + aCol]     = hi0;
        *(v8bf*)&sAhi[buf][aRow * BK + aCol + 8] = hi1;
        *(v8bf*)&sAlo[buf][aRow * BK + aCol]     = lo0;
        *(v8bf*)&sAlo[buf][aRow * BK + aCol + 8] = lo1;

        // ---- B: pre-split bf16 weights, straight 16B copies ----
        const size_t wOff = (size_t)(blockN0 + bRow) * K + (size_t)kt * BK + bCol;
        *(v8bf*)&sBhi[buf][bRow * BK + bCol]     = *(const v8bf*)(Whi + wOff);
        *(v8bf*)&sBhi[buf][bRow * BK + bCol + 8] = *(const v8bf*)(Whi + wOff + 8);
        *(v8bf*)&sBlo[buf][bRow * BK + bCol]     = *(const v8bf*)(Wlo + wOff);
        *(v8bf*)&sBlo[buf][bRow * BK + bCol + 8] = *(const v8bf*)(Wlo + wOff + 8);

        // ---- L2 prefetch for kt+2 (global_prefetch_b8) ----
        if (kt + 2 < nk) {
            __builtin_prefetch(src + 2 * BK, 0, 1);
            __builtin_prefetch(Whi + wOff + 2 * BK, 0, 1);
            __builtin_prefetch(Wlo + wOff + 2 * BK, 0, 1);
        }
    };

    auto compute = [&](int buf) {
        v16bf amh[MFRAG], aml[MFRAG], bnh[NFRAG], bnl[NFRAG];
#pragma unroll
        for (int mf = 0; mf < MFRAG; ++mf) {
            const int r = (mWave + mf * 16 + laneRow) * BK + laneHalf * 16;
            amh[mf] = *(const v16bf*)&sAhi[buf][r];
            aml[mf] = *(const v16bf*)&sAlo[buf][r];
        }
#pragma unroll
        for (int nf = 0; nf < NFRAG; ++nf) {
            const int r = (nWave + nf * 16 + laneRow) * BK + laneHalf * 16;
            bnh[nf] = *(const v16bf*)&sBhi[buf][r];
            bnl[nf] = *(const v16bf*)&sBlo[buf][r];
        }
#pragma unroll
        for (int mf = 0; mf < MFRAG; ++mf) {
#pragma unroll
            for (int nf = 0; nf < NFRAG; ++nf) {
                // 3xBF16 split: hi*hi + hi*lo + lo*hi (lo*lo dropped)
                acc[mf][nf] = __builtin_amdgcn_wmma_f32_16x16x32_bf16(
                    false, amh[mf], false, bnh[nf], (short)0, acc[mf][nf], false, false);
                acc[mf][nf] = __builtin_amdgcn_wmma_f32_16x16x32_bf16(
                    false, amh[mf], false, bnl[nf], (short)0, acc[mf][nf], false, false);
                acc[mf][nf] = __builtin_amdgcn_wmma_f32_16x16x32_bf16(
                    false, aml[mf], false, bnh[nf], (short)0, acc[mf][nf], false, false);
            }
        }
    };

    load_tile(0, 0);
    __syncthreads();
#pragma unroll 2
    for (int kt = 0; kt < nk; ++kt) {
        const int buf = kt & 1;
        if (kt + 1 < nk) load_tile(kt + 1, buf ^ 1);
        compute(buf);
        __syncthreads();
    }

    // Epilogue: C-frag layout (lane&15 -> N, lane>>4 -> M+8, vgpr e -> M+e)
#pragma unroll
    for (int mf = 0; mf < MFRAG; ++mf) {
#pragma unroll
        for (int nf = 0; nf < NFRAG; ++nf) {
            const int n  = blockN0 + nWave + nf * 16 + laneRow;
            const int m0 = blockM0 + mWave + mf * 16 + laneHalf * 8;
            const float bv = bias[n];
#pragma unroll
            for (int e = 0; e < 8; ++e) {
                out[(size_t)(m0 + e) * N + n] = acc[mf][nf][e] + bv;
            }
        }
    }
}

// ---------------------------------------------------------------------------
extern "C" void kernel_launch(void* const* d_in, const int* in_sizes, int n_in,
                              void* d_out, int out_size, void* d_ws, size_t ws_size,
                              hipStream_t stream) {
    const float* x  = (const float*)d_in[0];   // [4,4096,3072]
    const float* W  = (const float*)d_in[1];   // [3072,3072]
    const float* b  = (const float*)d_in[2];   // [3072]
    const float* lA = (const float*)d_in[3];   // [4,32,3072]
    const float* lB = (const float*)d_in[4];   // [4,3072,32]
    float* out = (float*)d_out;                // [4,4096,3072]

    // Workspace: bf16 hi/lo split of W_eff, 2 * 3072*3072 * 2B = 37.75 MB
    bf16_t* whi = (bf16_t*)d_ws;
    bf16_t* wlo = whi + (size_t)D_DIM * D_DIM;

    (void)in_sizes; (void)n_in; (void)out_size; (void)ws_size;

    // Fold LoRA into W and split to bf16 hi/lo.
    dim3 g1(D_DIM / 256, D_DIM);
    build_weff_kernel<<<g1, 256, 0, stream>>>(W, lA, lB, whi, wlo);

    // Single fused GEMM + bias. N fastest so W_eff (38 MB) stays L2-resident.
    dim3 g2(D_DIM / BN, M_DIM / BM);   // (24, 128)
    lora_gemm_kernel<<<g2, 256, 0, stream>>>(x, whi, wlo, b, out,
                                             M_DIM, D_DIM, D_DIM);
}
